// TransformerChessModel_64501818851656
// MI455X (gfx1250) — matrix operand order repaired and checked by reading, not compiled, not run on hardware
//
#include <hip/hip_runtime.h>

typedef __attribute__((ext_vector_type(16))) _Float16 v16h;
typedef __attribute__((ext_vector_type(8)))  float    v8f;

#define DMODEL 256
#define NHEAD  8
#define NLAYER 6
#define FFDIM  1024
#define NSQ    64
#define HDIM   32

// ---------------------------------------------------------------------------
// WMMA fragment loaders (16x16x32 f16, wave32; contiguous k = 16*(lane/16) + s in A and B)
// ---------------------------------------------------------------------------

// A fragment: 16x32 tile of row-major f16 matrix A (lda in elements).
__device__ inline v16h load_a_frag(const _Float16* A, int lda, int row0, int k0, int lane) {
  union { v16h h; unsigned u[8]; } f;
  int r    = row0 + (lane & 15);
  int koff = (lane >> 4) * 16;
  const _Float16* base = A + (size_t)r * lda + k0 + koff;
#pragma unroll
  for (int i = 0; i < 4; ++i) {
    f.u[i]     = *(const unsigned*)(base + 2 * i);
    f.u[i + 4] = *(const unsigned*)(base + 8 + 2 * i);
  }
  return f.h;
}

// B fragment where B[k][n] = W[n][k], W row-major [N,K] (x @ W^T).
__device__ inline v16h load_b_frag_wt(const _Float16* W, int ldw, int col0, int k0, int lane) {
  union { v16h h; unsigned u[8]; } f;
  int n  = col0 + (lane & 15);
  int kb = (lane >> 4) * 16;
  const _Float16* base = W + (size_t)n * ldw + k0 + kb;
#pragma unroll
  for (int i = 0; i < 8; ++i) f.u[i] = *(const unsigned*)(base + 2 * i);
  return f.h;
}

// B fragment where B[k][n] = V[k][n], V row-major [K,N] (ldv elements).
__device__ inline v16h load_b_frag_rm(const _Float16* V, int ldv, int col0, int k0, int lane) {
  union { v16h h; _Float16 e[16]; } f;
  int n  = col0 + (lane & 15);
  int kb = k0 + (lane >> 4) * 16;
#pragma unroll
  for (int i = 0; i < 8; ++i) {
    f.e[2 * i]     = V[(size_t)(kb + 2 * i) * ldv + n];
    f.e[2 * i + 1] = V[(size_t)(kb + 2 * i + 1) * ldv + n];
  }
  return f.h;
}

// ---------------------------------------------------------------------------
// Register-blocked GEMM: Y[M,N] = X[M,K] @ W[N,K]^T + bias, optional ReLU.
// 1 wave -> 32x64 output tile (2 A-frags x 4 B-frags -> 8 WMMAs / k-step).
// Block = 8 waves as 4(M) x 2(N): 128(M) x 128(N). grid = (M/128, N/128).
// bias must be non-null.
// ---------------------------------------------------------------------------
__global__ void __launch_bounds__(256)
gemm_wmma_kernel(const _Float16* __restrict__ A, int lda,
                 const _Float16* __restrict__ W, int ldw,
                 const float* __restrict__ bias,
                 float* __restrict__ Cf, _Float16* __restrict__ Ch, int ldc,
                 int M, int N, int K, int relu)
{
  int lane = threadIdx.x & 31;
  int wave = threadIdx.x >> 5;
  int wm = wave >> 1;            // 0..3
  int wn = wave & 1;             // 0..1
  int tm = blockIdx.x * 128 + wm * 32;
  int tn = blockIdx.y * 128 + wn * 64;
  if (tm >= M) return;

  v8f acc[2][4] = {};
  for (int k0 = 0; k0 < K; k0 += 32) {
    v16h af[2], bf[4];
#pragma unroll
    for (int mi = 0; mi < 2; ++mi)
      af[mi] = load_a_frag(A, lda, tm + mi * 16, k0, lane);
#pragma unroll
    for (int ni = 0; ni < 4; ++ni)
      bf[ni] = load_b_frag_wt(W, ldw, tn + ni * 16, k0, lane);
#pragma unroll
    for (int mi = 0; mi < 2; ++mi)
#pragma unroll
      for (int ni = 0; ni < 4; ++ni)
        acc[mi][ni] = __builtin_amdgcn_wmma_f32_16x16x32_f16(
            false, af[mi], false, bf[ni], (short)0, acc[mi][ni], false, false);
  }

  // bias + activation applied in-register (branch-free)
#pragma unroll
  for (int ni = 0; ni < 4; ++ni) {
    float bv = bias[tn + ni * 16 + (lane & 15)];
#pragma unroll
    for (int mi = 0; mi < 2; ++mi)
#pragma unroll
      for (int r = 0; r < 8; ++r) {
        float v = acc[mi][ni][r] + bv;
        acc[mi][ni][r] = relu ? fmaxf(v, 0.0f) : v;
      }
  }

  int col0  = tn + (lane & 15);
  int rofs  = (lane >> 4) << 3;
  if (Cf) {
#pragma unroll
    for (int mi = 0; mi < 2; ++mi)
#pragma unroll
      for (int ni = 0; ni < 4; ++ni)
#pragma unroll
        for (int r = 0; r < 8; ++r)
          Cf[(size_t)(tm + mi * 16 + rofs + r) * ldc + col0 + ni * 16] = acc[mi][ni][r];
  }
  if (Ch) {
#pragma unroll
    for (int mi = 0; mi < 2; ++mi)
#pragma unroll
      for (int ni = 0; ni < 4; ++ni)
#pragma unroll
        for (int r = 0; r < 8; ++r)
          Ch[(size_t)(tm + mi * 16 + rofs + r) * ldc + col0 + ni * 16] = (_Float16)acc[mi][ni][r];
  }
}

// ---------------------------------------------------------------------------
// Tokenize + embedding + positional encoding. 1 block = 1 token (B*64 blocks).
// ---------------------------------------------------------------------------
__global__ void __launch_bounds__(256)
embed_kernel(const float* __restrict__ board, const float* __restrict__ emb,
             float* __restrict__ x32, _Float16* __restrict__ xh)
{
  __shared__ int tok;
  int n = blockIdx.x;            // token = b*64 + s
  int b = n >> 6, s = n & 63;
  if (threadIdx.x == 0) {
    const float* cell = board + (size_t)b * (14 * 64) + s;   // [B,14,8,8]
    float mx = cell[0]; int best = 0;
    for (int c = 1; c < 14; ++c) {
      float v = cell[c * 64];
      if (v > mx) { mx = v; best = c; }
    }
    tok = (mx > 0.0f) ? best + 1 : 0;
  }
  __syncthreads();
  int d = threadIdx.x;
  float fi = (float)(s >> 3), fj = (float)(s & 7);
  float kf = (float)(d & ~1);
  float invdiv = powf(10000.0f, -kf * (1.0f / 256.0f));
  float pe = (d & 1) ? cosf(fj * invdiv) : sinf(fi * invdiv);
  float v = emb[tok * DMODEL + d] + pe;
  size_t idx = (size_t)n * DMODEL + d;
  x32[idx] = v;
  xh[idx]  = (_Float16)v;
}

// ---------------------------------------------------------------------------
// Fused attention for one (board, head). grid = (B, 8), block = 256.
// ---------------------------------------------------------------------------
__global__ void __launch_bounds__(256)
attention_kernel(const _Float16* __restrict__ qkv, _Float16* __restrict__ oh,
                 float scale)
{
  __shared__ float     sc[64][65];
  __shared__ _Float16  ph[64][64];
  int b = blockIdx.x, h = blockIdx.y;
  int lane = threadIdx.x & 31, wave = threadIdx.x >> 5;

  const _Float16* base = qkv + (size_t)b * NSQ * (3 * DMODEL);
  const _Float16* Q  = base + h * HDIM;
  const _Float16* Km = base + DMODEL + h * HDIM;
  const _Float16* Vm = base + 2 * DMODEL + h * HDIM;

  // scores: 16 tiles of 16x16, K = HD = 32 in a single WMMA step
  for (int t = wave; t < 16; t += 8) {
    int mi = t >> 2, ni = t & 3;
    v16h af = load_a_frag(Q, 3 * DMODEL, mi * 16, 0, lane);
    v16h bf = load_b_frag_wt(Km, 3 * DMODEL, ni * 16, 0, lane);
    v8f acc = {};
    acc = __builtin_amdgcn_wmma_f32_16x16x32_f16(false, af, false, bf,
                                                 (short)0, acc, false, false);
    int col = ni * 16 + (lane & 15);
    int rb  = mi * 16 + ((lane >> 4) << 3);
#pragma unroll
    for (int r = 0; r < 8; ++r) sc[rb + r][col] = acc[r] * scale;
  }
  __syncthreads();

  // softmax: one thread per row (S = 64)
  if (threadIdx.x < 64) {
    int row = threadIdx.x;
    float mx = sc[row][0];
    for (int c = 1; c < 64; ++c) mx = fmaxf(mx, sc[row][c]);
    float sum = 0.0f;
    for (int c = 0; c < 64; ++c) sum += __expf(sc[row][c] - mx);
    float inv = 1.0f / sum;
    for (int c = 0; c < 64; ++c)
      ph[row][c] = (_Float16)(__expf(sc[row][c] - mx) * inv);
  }
  __syncthreads();

  // O = P x V : 8 tiles (4 x M, 2 x N), one per wave, K = 64 in 2 steps
  {
    int mi = wave >> 1, ni = wave & 1;
    v8f acc = {};
    for (int k0 = 0; k0 < 64; k0 += 32) {
      v16h af = load_a_frag(&ph[0][0], 64, mi * 16, k0, lane);
      v16h bf = load_b_frag_rm(Vm, 3 * DMODEL, ni * 16, k0, lane);
      acc = __builtin_amdgcn_wmma_f32_16x16x32_f16(false, af, false, bf,
                                                   (short)0, acc, false, false);
    }
    int col = h * HDIM + ni * 16 + (lane & 15);
    int rb  = mi * 16 + ((lane >> 4) << 3);
#pragma unroll
    for (int r = 0; r < 8; ++r) {
      size_t row = (size_t)b * NSQ + rb + r;
      oh[row * DMODEL + col] = (_Float16)acc[r];
    }
  }
}

// ---------------------------------------------------------------------------
// x = LayerNorm(xin + y) * g + b   (post-norm). 1 block = 1 token row.
// ---------------------------------------------------------------------------
__global__ void __launch_bounds__(256)
add_ln_kernel(const float* __restrict__ xin, const float* __restrict__ y,
              const float* __restrict__ g, const float* __restrict__ bta,
              float* __restrict__ xout, _Float16* __restrict__ xh)
{
  __shared__ float red[256];
  int row = blockIdx.x, t = threadIdx.x;
  size_t idx = (size_t)row * DMODEL + t;
  float v = xin[idx] + y[idx];
  red[t] = v; __syncthreads();
  for (int s = 128; s > 0; s >>= 1) { if (t < s) red[t] += red[t + s]; __syncthreads(); }
  float mean = red[0] * (1.0f / 256.0f);
  __syncthreads();
  float d = v - mean;
  red[t] = d * d; __syncthreads();
  for (int s = 128; s > 0; s >>= 1) { if (t < s) red[t] += red[t + s]; __syncthreads(); }
  float var = red[0] * (1.0f / 256.0f);
  float o = d * rsqrtf(var + 1e-5f) * g[t] + bta[t];
  xout[idx] = o;
  xh[idx]   = (_Float16)o;
}

// ---------------------------------------------------------------------------
// pooled mean + value head (tanh) + classification head. 1 block = 1 board.
// ---------------------------------------------------------------------------
__global__ void __launch_bounds__(256)
heads_kernel(const float* __restrict__ x32,
             const float* __restrict__ vw1, const float* __restrict__ vb1,
             const float* __restrict__ vw2, const float* __restrict__ vb2,
             const float* __restrict__ cw1, const float* __restrict__ cb1,
             const float* __restrict__ cw2, const float* __restrict__ cb2,
             float* __restrict__ out_value, float* __restrict__ out_cls)
{
  __shared__ float pl[256];
  __shared__ float vh[256];
  __shared__ float ch[256];
  __shared__ float red[256];
  int b = blockIdx.x, t = threadIdx.x;
  const float* xb = x32 + (size_t)b * NSQ * DMODEL;
  float sum = 0.0f;
  for (int s = 0; s < NSQ; ++s) sum += xb[s * DMODEL + t];
  pl[t] = sum * (1.0f / 64.0f);
  __syncthreads();
  float a1 = vb1[t], a2 = cb1[t];
  for (int k = 0; k < 256; ++k) {
    float p = pl[k];
    a1 += p * vw1[t * 256 + k];
    a2 += p * cw1[t * 256 + k];
  }
  vh[t] = fmaxf(a1, 0.0f);
  ch[t] = fmaxf(a2, 0.0f);
  __syncthreads();
  red[t] = vh[t] * vw2[t]; __syncthreads();
  for (int s = 128; s > 0; s >>= 1) { if (t < s) red[t] += red[t + s]; __syncthreads(); }
  if (t == 0) out_value[b] = tanhf(red[0] + vb2[0]);
  for (int k = 0; k < 3; ++k) {
    __syncthreads();
    red[t] = ch[t] * cw2[k * 256 + t]; __syncthreads();
    for (int s = 128; s > 0; s >>= 1) { if (t < s) red[t] += red[t + s]; __syncthreads(); }
    if (t == 0) out_cls[(size_t)b * 3 + k] = red[0] + cb2[k];
  }
}

// ---------------------------------------------------------------------------
__global__ void f32_to_f16_kernel(const float* __restrict__ src,
                                  _Float16* __restrict__ dst, size_t n)
{
  size_t i      = (size_t)blockIdx.x * blockDim.x + threadIdx.x;
  size_t stride = (size_t)gridDim.x * blockDim.x;
  for (; i < n; i += stride) dst[i] = (_Float16)src[i];
}

// ---------------------------------------------------------------------------
extern "C" void kernel_launch(void* const* d_in, const int* in_sizes, int n_in,
                              void* d_out, int out_size, void* d_ws, size_t ws_size,
                              hipStream_t stream)
{
  const float* board  = (const float*)d_in[0];
  const float* emb    = (const float*)d_in[1];
  const float* qkv_w  = (const float*)d_in[2];
  const float* qkv_b  = (const float*)d_in[3];
  const float* out_w  = (const float*)d_in[4];
  const float* out_b  = (const float*)d_in[5];
  const float* ln1_g  = (const float*)d_in[6];
  const float* ln1_b  = (const float*)d_in[7];
  const float* ln2_g  = (const float*)d_in[8];
  const float* ln2_b  = (const float*)d_in[9];
  const float* ff1_w  = (const float*)d_in[10];
  const float* ff1_b  = (const float*)d_in[11];
  const float* ff2_w  = (const float*)d_in[12];
  const float* ff2_b  = (const float*)d_in[13];
  const float* vw1    = (const float*)d_in[14];
  const float* vb1    = (const float*)d_in[15];
  const float* vw2    = (const float*)d_in[16];
  const float* vb2    = (const float*)d_in[17];
  const float* pw1    = (const float*)d_in[18];
  const float* pb1    = (const float*)d_in[19];
  const float* pw2    = (const float*)d_in[20];
  const float* pb2    = (const float*)d_in[21];
  const float* cw1    = (const float*)d_in[22];
  const float* cb1    = (const float*)d_in[23];
  const float* cw2    = (const float*)d_in[24];
  const float* cb2    = (const float*)d_in[25];

  const int B = in_sizes[0] / (14 * 64);   // 4096
  const size_t N = (size_t)B * NSQ;        // token rows

  // ---- workspace layout -------------------------------------------------
  size_t off = 0;
  auto alloc = [&](size_t bytes) {
    size_t o = off;
    off += (bytes + 255) & ~(size_t)255;
    return o;
  };
  char* ws = (char*)d_ws;

  _Float16* qkvw_h = (_Float16*)(ws + alloc(sizeof(_Float16) * NLAYER * 3 * DMODEL * DMODEL));
  _Float16* outw_h = (_Float16*)(ws + alloc(sizeof(_Float16) * NLAYER * DMODEL * DMODEL));
  _Float16* ff1w_h = (_Float16*)(ws + alloc(sizeof(_Float16) * NLAYER * FFDIM * DMODEL));
  _Float16* ff2w_h = (_Float16*)(ws + alloc(sizeof(_Float16) * NLAYER * DMODEL * FFDIM));
  _Float16* pw1_h  = (_Float16*)(ws + alloc(sizeof(_Float16) * 1024 * (DMODEL * NSQ)));
  _Float16* pw2_h  = (_Float16*)(ws + alloc(sizeof(_Float16) * 4096 * 1024));

  _Float16* xh   = (_Float16*)(ws + alloc(sizeof(_Float16) * N * DMODEL));
  float*    x32  = (float*)   (ws + alloc(sizeof(float)    * N * DMODEL));
  float*    y32  = (float*)   (ws + alloc(sizeof(float)    * N * DMODEL));
  // big region doubles as (qkv + attn-out) during attention, and ff-hidden
  _Float16* big  = (_Float16*)(ws + alloc(sizeof(_Float16) * N * FFDIM));
  _Float16* qkvh = big;                        // [N, 768]
  _Float16* oh   = big + N * (3 * DMODEL);     // [N, 256]
  _Float16* ffh  = big;                        // [N, 1024]
  _Float16* ph1  = (_Float16*)(ws + alloc(sizeof(_Float16) * (size_t)B * 1024));

  float* out_value = (float*)d_out;
  float* out_pol   = (float*)d_out + B;
  float* out_cls   = (float*)d_out + (size_t)B + (size_t)B * 4096;

  // ---- weight conversion f32 -> f16 ------------------------------------
  auto conv = [&](const float* s, _Float16* d, size_t n) {
    int blocks = (int)((n + 255) / 256);
    if (blocks > 65536) blocks = 65536;
    f32_to_f16_kernel<<<blocks, 256, 0, stream>>>(s, d, n);
  };
  conv(qkv_w, qkvw_h, (size_t)NLAYER * 3 * DMODEL * DMODEL);
  conv(out_w, outw_h, (size_t)NLAYER * DMODEL * DMODEL);
  conv(ff1_w, ff1w_h, (size_t)NLAYER * FFDIM * DMODEL);
  conv(ff2_w, ff2w_h, (size_t)NLAYER * DMODEL * FFDIM);
  conv(pw1,   pw1_h,  (size_t)1024 * DMODEL * NSQ);
  conv(pw2,   pw2_h,  (size_t)4096 * 1024);

  // ---- embedding --------------------------------------------------------
  embed_kernel<<<(int)N, 256, 0, stream>>>(board, emb, x32, xh);

  const float scale = 1.0f / sqrtf((float)HDIM);
  const int MB = (int)(N / 128);   // M super-tiles for token-row GEMMs

  // ---- transformer layers ----------------------------------------------
  for (int l = 0; l < NLAYER; ++l) {
    // QKV: [N,256] x [768,256]^T -> qkvh f16
    gemm_wmma_kernel<<<dim3(MB, (3 * DMODEL) / 128), 256, 0, stream>>>(
        xh, DMODEL, qkvw_h + (size_t)l * 3 * DMODEL * DMODEL, DMODEL,
        qkv_b + (size_t)l * 3 * DMODEL, nullptr, qkvh, 3 * DMODEL,
        (int)N, 3 * DMODEL, DMODEL, 0);

    // fused attention -> oh f16
    attention_kernel<<<dim3(B, NHEAD), 256, 0, stream>>>(qkvh, oh, scale);

    // out-proj: [N,256] x [256,256]^T -> y32 f32
    gemm_wmma_kernel<<<dim3(MB, DMODEL / 128), 256, 0, stream>>>(
        oh, DMODEL, outw_h + (size_t)l * DMODEL * DMODEL, DMODEL,
        out_b + (size_t)l * DMODEL, y32, nullptr, DMODEL,
        (int)N, DMODEL, DMODEL, 0);

    // x = LN(x + y)
    add_ln_kernel<<<(int)N, 256, 0, stream>>>(
        x32, y32, ln1_g + (size_t)l * DMODEL, ln1_b + (size_t)l * DMODEL, x32, xh);

    // FF1: relu([N,256] x [1024,256]^T) -> ffh f16
    gemm_wmma_kernel<<<dim3(MB, FFDIM / 128), 256, 0, stream>>>(
        xh, DMODEL, ff1w_h + (size_t)l * FFDIM * DMODEL, DMODEL,
        ff1_b + (size_t)l * FFDIM, nullptr, ffh, FFDIM,
        (int)N, FFDIM, DMODEL, 1);

    // FF2: [N,1024] x [256,1024]^T -> y32 f32
    gemm_wmma_kernel<<<dim3(MB, DMODEL / 128), 256, 0, stream>>>(
        ffh, FFDIM, ff2w_h + (size_t)l * DMODEL * FFDIM, FFDIM,
        ff2_b + (size_t)l * DMODEL, y32, nullptr, DMODEL,
        (int)N, DMODEL, FFDIM, 0);

    // x = LN(x + ff)
    add_ln_kernel<<<(int)N, 256, 0, stream>>>(
        x32, y32, ln2_g + (size_t)l * DMODEL, ln2_b + (size_t)l * DMODEL, x32, xh);
  }

  // ---- value + classification heads ------------------------------------
  heads_kernel<<<B, 256, 0, stream>>>(x32, vw1, vb1, vw2, vb2,
                                      cw1, cb1, cw2, cb2, out_value, out_cls);

  // ---- policy head ------------------------------------------------------
  // relu([B,16384] x [1024,16384]^T) -> ph1 f16
  gemm_wmma_kernel<<<dim3(B / 128, 1024 / 128), 256, 0, stream>>>(
      xh, DMODEL * NSQ, pw1_h, DMODEL * NSQ, pb1, nullptr, ph1, 1024,
      B, 1024, DMODEL * NSQ, 1);
  // [B,1024] x [4096,1024]^T -> policy f32 (directly into d_out)
  gemm_wmma_kernel<<<dim3(B / 128, 4096 / 128), 256, 0, stream>>>(
      ph1, 1024, pw2_h, 1024, pb2, out_pol, nullptr, 4096,
      B, 4096, 1024, 0);
}
